// AttnDecoderRNN_1434519077453
// MI455X (gfx1250) — matrix operand order repaired and checked by reading, not compiled, not run on hardware
//
#include <hip/hip_runtime.h>
#include <math.h>

// ---------------- problem constants ----------------
#define BATCH 128
#define HDIM  512
#define EDIM  256
#define LLEN  400
#define TLEN  100
#define VSZ   50000
#define G3    1536   // 3*H
#define NTILE 64     // GEMM block N-tile
#define NKSTEP (G3 / 32)

typedef __attribute__((ext_vector_type(16))) __bf16 v16bf;
typedef __attribute__((ext_vector_type(8)))  float  v8f;

__device__ __forceinline__ unsigned short f2bf(float f) {
    unsigned u = __float_as_uint(f);
    unsigned r = u + 0x7FFFu + ((u >> 16) & 1u);   // round-to-nearest-even
    return (unsigned short)(r >> 16);
}

// ================= Kernel 1: embed + GRU + te/td ==================
// one block per batch row, 512 threads
__global__ __launch_bounds__(512) void gru_kernel(
    const int* __restrict__ inputs, const float* __restrict__ hidden,
    const float* __restrict__ emb,
    const float* __restrict__ W_ih, const float* __restrict__ W_hh,
    const float* __restrict__ b_ih, const float* __restrict__ b_hh,
    const float* __restrict__ W_ae, const float* __restrict__ b_ae,
    const float* __restrict__ W_ad, const float* __restrict__ b_ad,
    float* __restrict__ h_new, float* __restrict__ te, float* __restrict__ td,
    float* __restrict__ h_out)
{
    __shared__ __align__(16) float sx[EDIM];
    __shared__ __align__(16) float sh[HDIM];
    __shared__ __align__(16) float sgi[G3];
    __shared__ __align__(16) float sgh[G3];
    __shared__ __align__(16) float shn[HDIM];

    const int b = blockIdx.x;
    const int tid = threadIdx.x;
    const int tok = inputs[b];
    if (tid < EDIM) sx[tid] = emb[(size_t)tok * EDIM + tid];
    sh[tid] = hidden[b * HDIM + tid];
    __syncthreads();

    for (int g = tid; g < G3; g += 512) {
        const float4* wi = (const float4*)(W_ih + (size_t)g * EDIM);
        float si = b_ih[g];
        #pragma unroll 4
        for (int i = 0; i < EDIM / 4; ++i) {
            float4 w = wi[i]; float4 x = ((const float4*)sx)[i];
            si += w.x * x.x + w.y * x.y + w.z * x.z + w.w * x.w;
        }
        sgi[g] = si;
        const float4* wh = (const float4*)(W_hh + (size_t)g * HDIM);
        float sg = b_hh[g];
        #pragma unroll 4
        for (int i = 0; i < HDIM / 4; ++i) {
            float4 w = wh[i]; float4 h4 = ((const float4*)sh)[i];
            sg += w.x * h4.x + w.y * h4.y + w.z * h4.z + w.w * h4.w;
        }
        sgh[g] = sg;
    }
    __syncthreads();

    // gates: [r | z | n] each HDIM wide
    float r = 1.f / (1.f + expf(-(sgi[tid] + sgh[tid])));
    float z = 1.f / (1.f + expf(-(sgi[HDIM + tid] + sgh[HDIM + tid])));
    float n = tanhf(sgi[2 * HDIM + tid] + r * sgh[2 * HDIM + tid]);
    float hn = (1.f - z) * n + z * sh[tid];
    shn[tid] = hn;
    h_new[b * HDIM + tid] = hn;
    h_out[b * HDIM + tid] = hn;
    __syncthreads();

    const float4* wa = (const float4*)(W_ae + (size_t)tid * HDIM);
    const float4* wd = (const float4*)(W_ad + (size_t)tid * HDIM);
    float sa = b_ae[tid], sd = b_ad[tid];
    #pragma unroll 4
    for (int i = 0; i < HDIM / 4; ++i) {
        float4 a = wa[i], d = wd[i], h4 = ((const float4*)shn)[i];
        sa += a.x * h4.x + a.y * h4.y + a.z * h4.z + a.w * h4.w;
        sd += d.x * h4.x + d.y * h4.y + d.z * h4.z + d.w * h4.w;
    }
    te[b * HDIM + tid] = sa;
    td[b * HDIM + tid] = sd;
}

// ================= Kernel 2: attention scores (one wave per dot) ==========
__global__ __launch_bounds__(256) void scores_kernel(
    const float* __restrict__ te, const float* __restrict__ td,
    const float* __restrict__ enc, const float* __restrict__ dec,
    float* __restrict__ et, float* __restrict__ etd)
{
    const int wid  = blockIdx.x * (blockDim.x >> 5) + (threadIdx.x >> 5);
    const int lane = threadIdx.x & 31;
    float s = 0.f;
    if (wid < BATCH * LLEN) {
        const int b = wid & (BATCH - 1), l = wid >> 7;
        const float4* v = (const float4*)(enc + ((size_t)l * BATCH + b) * HDIM);
        const float4* q = (const float4*)(te + b * HDIM);
        #pragma unroll
        for (int i = 0; i < 4; ++i) {
            float4 a = v[lane + 32 * i], c = q[lane + 32 * i];
            s += a.x * c.x + a.y * c.y + a.z * c.z + a.w * c.w;
        }
        #pragma unroll
        for (int o = 16; o > 0; o >>= 1) s += __shfl_down(s, o, 32);
        if (lane == 0) et[b * LLEN + l] = s;
    } else if (wid < BATCH * LLEN + BATCH * TLEN) {
        const int w2 = wid - BATCH * LLEN;
        const int b = w2 & (BATCH - 1), t = w2 >> 7;
        const float4* v = (const float4*)(dec + ((size_t)t * BATCH + b) * HDIM);
        const float4* q = (const float4*)(td + b * HDIM);
        #pragma unroll
        for (int i = 0; i < 4; ++i) {
            float4 a = v[lane + 32 * i], c = q[lane + 32 * i];
            s += a.x * c.x + a.y * c.y + a.z * c.z + a.w * c.w;
        }
        #pragma unroll
        for (int o = 16; o > 0; o >>= 1) s += __shfl_down(s, o, 32);
        if (lane == 0) etd[b * TLEN + t] = s;
    }
}

// ================= Kernel 3: softmaxes =================
__global__ __launch_bounds__(512) void softmax_kernel(
    const float* __restrict__ et, const float* __restrict__ etd,
    const int* __restrict__ enc_len,
    float* __restrict__ alpha_e, float* __restrict__ alpha_d)
{
    const int b = blockIdx.x, tid = threadIdx.x;
    __shared__ float red[512];

    // encoder softmax (L=400)
    float val = -INFINITY;
    if (tid < LLEN) {
        val = et[b * LLEN + tid];
        int len = enc_len[b];
        if (LLEN < len && tid >= len) val = -INFINITY;   // faithful (always false)
    }
    red[tid] = val; __syncthreads();
    for (int s = 256; s > 0; s >>= 1) { if (tid < s) red[tid] = fmaxf(red[tid], red[tid + s]); __syncthreads(); }
    float mx = red[0]; __syncthreads();
    float e = (tid < LLEN) ? expf(val - mx) : 0.f;
    red[tid] = e; __syncthreads();
    for (int s = 256; s > 0; s >>= 1) { if (tid < s) red[tid] += red[tid + s]; __syncthreads(); }
    float inv = 1.f / red[0]; __syncthreads();
    if (tid < LLEN) alpha_e[b * LLEN + tid] = e * inv;

    // decoder softmax (T=100)
    float v2 = (tid < TLEN) ? etd[b * TLEN + tid] : -INFINITY;
    red[tid] = v2; __syncthreads();
    for (int s = 256; s > 0; s >>= 1) { if (tid < s) red[tid] = fmaxf(red[tid], red[tid + s]); __syncthreads(); }
    float mx2 = red[0]; __syncthreads();
    float e2 = (tid < TLEN) ? expf(v2 - mx2) : 0.f;
    red[tid] = e2; __syncthreads();
    for (int s = 256; s > 0; s >>= 1) { if (tid < s) red[tid] += red[tid + s]; __syncthreads(); }
    float inv2 = 1.f / red[0];
    if (tid < TLEN) alpha_d[b * TLEN + tid] = e2 * inv2;
}

// ======= Kernel 4: contexts + assemble new_hidden (f32 + bf16) =======
__global__ __launch_bounds__(512) void context_kernel(
    const float* __restrict__ alpha_e, const float* __restrict__ alpha_d,
    const float* __restrict__ enc, const float* __restrict__ dec,
    const float* __restrict__ h_new,
    float* __restrict__ nh, unsigned short* __restrict__ Abf)
{
    const int b = blockIdx.x, j = threadIdx.x;
    __shared__ float sae[LLEN];
    __shared__ float sad[TLEN];
    if (j < LLEN) sae[j] = alpha_e[b * LLEN + j];
    if (j < TLEN) sad[j] = alpha_d[b * TLEN + j];
    __syncthreads();

    float ce = 0.f;
    for (int l = 0; l < LLEN; ++l) ce += sae[l] * enc[((size_t)l * BATCH + b) * HDIM + j];
    float cd = 0.f;
    for (int t = 0; t < TLEN; ++t) cd += sad[t] * dec[((size_t)t * BATCH + b) * HDIM + j];
    float hv = h_new[b * HDIM + j];

    const size_t base = (size_t)b * G3;
    nh[base + j] = ce; nh[base + HDIM + j] = hv; nh[base + 2 * HDIM + j] = cd;
    Abf[base + j] = f2bf(ce); Abf[base + HDIM + j] = f2bf(hv); Abf[base + 2 * HDIM + j] = f2bf(cd);
}

// ================= Kernel 5: pointer prob (fp32, exact) =================
__global__ void pointer_kernel(const float* __restrict__ nh,
                               const float* __restrict__ W_ptr,
                               const float* __restrict__ b_ptr,
                               float* __restrict__ p)
{
    const int b = threadIdx.x;   // 128 threads, 1 block
    float s = b_ptr[0];
    const float4* w = (const float4*)W_ptr;
    const float4* x = (const float4*)(nh + (size_t)b * G3);
    #pragma unroll 4
    for (int i = 0; i < G3 / 4; ++i) {
        float4 a = w[i], c = x[i];
        s += a.x * c.x + a.y * c.y + a.z * c.z + a.w * c.w;
    }
    p[b] = s;
}

// ======= Kernel 6: big GEMM via v_wmma_f32_16x16x32_bf16 =======
// logits[b,v] = sum_k nh_bf16[b,k] * bf16(W_out[v,k]) + b_out[v]
// block tile M=128 x N=64, 8 waves: wave w -> M-tile w, 4 N-tiles each
// 2-deep LDS double buffer: global loads for tile i+1 overlap WMMAs on tile i;
// single workgroup barrier per K-step.
__global__ __launch_bounds__(256) void out_gemm_kernel(
    const unsigned short* __restrict__ Abf,
    const float* __restrict__ W_out, const float* __restrict__ b_out,
    float* __restrict__ logits)
{
    __shared__ __align__(16) unsigned short ALds[2][BATCH * 32]; // 2 x 128x32 bf16
    __shared__ __align__(16) unsigned short BLds[2][NTILE * 32]; // 2 x 64n x 32k bf16

    const int tid  = threadIdx.x;
    const int wave = tid >> 5, lane = tid & 31;
    const int vbase = blockIdx.x * NTILE;
    const int khalf = lane >> 4;       // 0 or 1
    const int m15   = lane & 15;

    // loader roles
    const int arow = tid >> 1, achunk = tid & 1;     // A: 16 halves / thread
    const unsigned short* aptr = Abf + (size_t)arow * G3 + achunk * 16;
    const int bn = tid >> 2, bkg = tid & 3;          // B: 8 floats / thread
    const int bv = vbase + bn;
    const float* bptr = (bv < VSZ) ? (W_out + (size_t)bv * G3 + bkg * 8) : (const float*)0;

    union Frag { v16bf v; unsigned u[8]; };
    v8f acc[4] = {v8f{}, v8f{}, v8f{}, v8f{}};

    uint4  areg0, areg1;
    float4 breg0 = make_float4(0.f, 0.f, 0.f, 0.f);
    float4 breg1 = make_float4(0.f, 0.f, 0.f, 0.f);

    // ---- prologue: fetch + stage tile 0 into buffer 0 ----
    areg0 = ((const uint4*)aptr)[0];
    areg1 = ((const uint4*)aptr)[1];
    if (bptr) { breg0 = ((const float4*)bptr)[0]; breg1 = ((const float4*)bptr)[1]; }
    *(uint4*)&ALds[0][arow * 32 + achunk * 16]     = areg0;
    *(uint4*)&ALds[0][arow * 32 + achunk * 16 + 8] = areg1;
    {
        uint4 bu;
        bu.x = (unsigned)f2bf(breg0.x) | ((unsigned)f2bf(breg0.y) << 16);
        bu.y = (unsigned)f2bf(breg0.z) | ((unsigned)f2bf(breg0.w) << 16);
        bu.z = (unsigned)f2bf(breg1.x) | ((unsigned)f2bf(breg1.y) << 16);
        bu.w = (unsigned)f2bf(breg1.z) | ((unsigned)f2bf(breg1.w) << 16);
        *(uint4*)&BLds[0][bn * 32 + bkg * 8] = bu;
    }
    __syncthreads();

    for (int it = 0; it < NKSTEP; ++it) {
        const int cur = it & 1, nxt = cur ^ 1;
        const int k0n = (it + 1) * 32;

        // issue next tile's global loads early (overlap with WMMA below)
        if (it + 1 < NKSTEP) {
            const unsigned short* ap = aptr + k0n;
            areg0 = ((const uint4*)ap)[0];
            areg1 = ((const uint4*)ap)[1];
            if (bptr) {
                const float* bp = bptr + k0n;
                breg0 = ((const float4*)bp)[0];
                breg1 = ((const float4*)bp)[1];
                if (k0n + 32 < G3) __builtin_prefetch(bptr + k0n + 32, 0, 1);
            }
        }

        // ---- compute on buffer `cur` ----
        Frag a;
        const int am = (wave << 4) + m15;
        #pragma unroll
        for (int r = 0; r < 8; ++r) {   // A 16x32 bf16 lane layout (ISA 7.12.2)
            int kk = ((r < 4) ? (r << 1) : (16 + ((r - 4) << 1))) + (khalf << 3);
            a.u[r] = *(const unsigned*)&ALds[cur][am * 32 + kk];
        }
        #pragma unroll
        for (int nt = 0; nt < 4; ++nt) {
            Frag bf;
            const int nn = nt * 16 + m15;   // B 32x16: same k order as A (lanes 0-15 K0..7,16..23; 16-31 K8..15,24..31)
            #pragma unroll
            for (int r = 0; r < 8; ++r) {
                int kk = ((r < 4) ? (r << 1) : (16 + ((r - 4) << 1))) + (khalf << 3);
                bf.u[r] = *(const unsigned*)&BLds[cur][nn * 32 + kk];
            }
            acc[nt] = __builtin_amdgcn_wmma_f32_16x16x32_bf16(false, a.v, false, bf.v,
                                                              (short)0, acc[nt], false, false);
        }

        // ---- stage next tile into buffer `nxt` (no conflict with `cur` readers) ----
        if (it + 1 < NKSTEP) {
            *(uint4*)&ALds[nxt][arow * 32 + achunk * 16]     = areg0;
            *(uint4*)&ALds[nxt][arow * 32 + achunk * 16 + 8] = areg1;
            uint4 bu;
            bu.x = (unsigned)f2bf(breg0.x) | ((unsigned)f2bf(breg0.y) << 16);
            bu.y = (unsigned)f2bf(breg0.z) | ((unsigned)f2bf(breg0.w) << 16);
            bu.z = (unsigned)f2bf(breg1.x) | ((unsigned)f2bf(breg1.y) << 16);
            bu.w = (unsigned)f2bf(breg1.z) | ((unsigned)f2bf(breg1.w) << 16);
            *(uint4*)&BLds[nxt][bn * 32 + bkg * 8] = bu;
        }
        __syncthreads();   // one barrier per K-step
    }

    // ---- epilogue: C/D layout M = r + 8*(lane>=16), N = lane%16 ----
    const int mb = (wave << 4) + (khalf << 3);
    #pragma unroll
    for (int nt = 0; nt < 4; ++nt) {
        const int v = vbase + nt * 16 + m15;
        if (v < VSZ) {
            const float bo = b_out[v];
            #pragma unroll
            for (int r = 0; r < 8; ++r)
                logits[(size_t)(mb + r) * VSZ + v] = acc[nt][r] + bo;
        }
    }
}

// ================= Kernel 7: per-row logsumexp over V =================
__global__ __launch_bounds__(256) void lse_kernel(const float* __restrict__ logits,
                                                  float* __restrict__ lse)
{
    const int b = blockIdx.x, tid = threadIdx.x;
    __shared__ float red[256];
    const float* row = logits + (size_t)b * VSZ;
    float m = -INFINITY;
    for (int i = tid; i < VSZ; i += 256) m = fmaxf(m, row[i]);
    red[tid] = m; __syncthreads();
    for (int s = 128; s > 0; s >>= 1) { if (tid < s) red[tid] = fmaxf(red[tid], red[tid + s]); __syncthreads(); }
    m = red[0]; __syncthreads();
    float sum = 0.f;
    for (int i = tid; i < VSZ; i += 256) sum += expf(row[i] - m);
    red[tid] = sum; __syncthreads();
    for (int s = 128; s > 0; s >>= 1) { if (tid < s) red[tid] += red[tid + s]; __syncthreads(); }
    if (tid == 0) lse[b] = m + logf(red[0]);
}

// ============ Kernel 8: final = (1-p)*(logit - lse), in-place ============
__global__ void final_kernel(float* __restrict__ out, const float* __restrict__ p,
                             const float* __restrict__ lse)
{
    const size_t i = (size_t)blockIdx.x * blockDim.x + threadIdx.x;
    if (i >= (size_t)BATCH * VSZ) return;
    const int b = (int)(i / VSZ);
    out[i] = (1.f - p[b]) * (out[i] - lse[b]);
}

// ====== Kernel 9: pointer scatter — deterministic, row-owner threads ======
__global__ void scatter_kernel(float* __restrict__ out, const float* __restrict__ p,
                               const float* __restrict__ alpha_e,
                               const int* __restrict__ enc_inputs)
{
    const int b = threadIdx.x;   // 128 threads, each owns output row b -> no races
    const float pb = p[b];
    for (int l = 0; l < LLEN; ++l) {
        const int idx = enc_inputs[l * BATCH + b];
        out[(size_t)b * VSZ + idx] += pb * alpha_e[b * LLEN + l];
    }
}

// ================= misc copy =================
__global__ void copy_kernel(const float* __restrict__ src, float* __restrict__ dst, int n)
{
    const int i = blockIdx.x * blockDim.x + threadIdx.x;
    if (i < n) dst[i] = src[i];
}

extern "C" void kernel_launch(void* const* d_in, const int* in_sizes, int n_in,
                              void* d_out, int out_size, void* d_ws, size_t ws_size,
                              hipStream_t stream)
{
    (void)in_sizes; (void)n_in; (void)out_size; (void)ws_size;
    const int*   inputs     = (const int*)  d_in[0];
    const float* hidden     = (const float*)d_in[1];
    const float* enc        = (const float*)d_in[2];
    const int*   enc_len    = (const int*)  d_in[3];
    const float* prevw      = (const float*)d_in[4];
    const float* dec        = (const float*)d_in[5];
    const int*   enc_inputs = (const int*)  d_in[6];
    const float* emb        = (const float*)d_in[8];
    const float* W_ih       = (const float*)d_in[9];
    const float* W_hh       = (const float*)d_in[10];
    const float* b_ih       = (const float*)d_in[11];
    const float* b_hh       = (const float*)d_in[12];
    const float* W_ae       = (const float*)d_in[13];
    const float* b_ae       = (const float*)d_in[14];
    const float* W_ad       = (const float*)d_in[15];
    const float* b_ad       = (const float*)d_in[16];
    const float* W_ptr      = (const float*)d_in[17];
    const float* b_ptr      = (const float*)d_in[18];
    const float* W_out      = (const float*)d_in[19];
    const float* b_out      = (const float*)d_in[20];

    float* out = (float*)d_out;
    float* h_out  = out + (size_t)BATCH * VSZ;            // (1,B,H)
    float* pw_out = h_out + BATCH * HDIM;                 // (B,1,L)

    // workspace layout (floats)
    float* ws       = (float*)d_ws;
    float* h_new    = ws;                                 // 65536
    float* te_      = ws + 65536;                         // 65536
    float* td_      = ws + 131072;                        // 65536
    float* e_t      = ws + 196608;                        // 51200
    float* alpha_e  = ws + 247808;                        // 51200
    float* e_td     = ws + 299008;                        // 12800
    float* alpha_d  = ws + 311808;                        // 12800
    float* nh       = ws + 324608;                        // 196608
    float* p_       = ws + 521216;                        // 128
    float* lse_     = ws + 521344;                        // 128
    unsigned short* Abf = (unsigned short*)(ws + 521472); // 196608 bf16

    gru_kernel<<<BATCH, 512, 0, stream>>>(inputs, hidden, emb, W_ih, W_hh, b_ih, b_hh,
                                          W_ae, b_ae, W_ad, b_ad, h_new, te_, td_, h_out);
    scores_kernel<<<(BATCH * LLEN + BATCH * TLEN + 7) / 8, 256, 0, stream>>>(
        te_, td_, enc, dec, e_t, e_td);
    softmax_kernel<<<BATCH, 512, 0, stream>>>(e_t, e_td, enc_len, alpha_e, alpha_d);
    context_kernel<<<BATCH, 512, 0, stream>>>(alpha_e, alpha_d, enc, dec, h_new, nh, Abf);
    pointer_kernel<<<1, BATCH, 0, stream>>>(nh, W_ptr, b_ptr, p_);
    copy_kernel<<<(BATCH * LLEN + 255) / 256, 256, 0, stream>>>(prevw, pw_out, BATCH * LLEN);

    out_gemm_kernel<<<(VSZ + NTILE - 1) / NTILE, 256, 0, stream>>>(Abf, W_out, b_out, out);
    lse_kernel<<<BATCH, 256, 0, stream>>>(out, lse_);
    final_kernel<<<(BATCH * VSZ + 255) / 256, 256, 0, stream>>>(out, p_, lse_);
    scatter_kernel<<<1, BATCH, 0, stream>>>(out, p_, alpha_e, enc_inputs);
}
